// Graph_resnet_24532853195294
// MI455X (gfx1250) — matrix operand order repaired and checked by reading, not compiled, not run on hardware
//
#include <hip/hip_runtime.h>

#define NNODES 100000
#define NEDGES 1200000

typedef __attribute__((ext_vector_type(16))) __bf16 v16bf;
typedef __attribute__((ext_vector_type(8)))  float  v8f;

// ---------------- elementwise helpers ----------------
__global__ void k_zero(float* __restrict__ p, long long n) {
  long long i = blockIdx.x * 256LL + threadIdx.x;
  if (i < n) p[i] = 0.f;
}

__global__ void k_copy(float* __restrict__ d, const float* __restrict__ s, long long n) {
  long long i = blockIdx.x * 256LL + threadIdx.x;
  if (i < n) d[i] = s[i];
}

__global__ void k_bias_init(float* __restrict__ out, const float* __restrict__ bias,
                            int Fout, long long n) {
  long long i = blockIdx.x * 256LL + threadIdx.x;
  if (i < n) out[i] = bias[i % Fout];
}

// T_prev <- 2*P - T_prev   (Chebyshev recurrence, in place)
__global__ void k_combine(float* __restrict__ tp, const float* __restrict__ p, long long n) {
  long long i = blockIdx.x * 256LL + threadIdx.x;
  if (i < n) tp[i] = 2.f * p[i] - tp[i];
}

// h <- relu(outB) + outS
__global__ void k_relu_skip(float* __restrict__ h, const float* __restrict__ ob,
                            const float* __restrict__ os, long long n) {
  long long i = blockIdx.x * 256LL + threadIdx.x;
  if (i < n) h[i] = fmaxf(ob[i], 0.f) + os[i];
}

// hcat = [h(64) | x(96)], row stride 160
__global__ void k_concat(float* __restrict__ hc, const float* __restrict__ h,
                         const float* __restrict__ x, int n) {
  long long i = blockIdx.x * 256LL + threadIdx.x;
  if (i >= (long long)n * 160) return;
  int node = (int)(i / 160);
  int j = (int)(i % 160);
  hc[i] = (j < 64) ? h[(size_t)node * 64 + j] : x[(size_t)node * 96 + (j - 64)];
}

// ---------------- edge-weight setup ----------------
__global__ void k_deg(float* __restrict__ deg, const int* __restrict__ src, int e) {
  int i = blockIdx.x * 256 + threadIdx.x;
  if (i < e) atomicAdd(deg + src[i], 1.f);
}

__global__ void k_dinv(float* __restrict__ d, int n) {
  int i = blockIdx.x * 256 + threadIdx.x;
  if (i < n) {
    float v = d[i];
    d[i] = (v > 0.f) ? rsqrtf(fmaxf(v, 1.f)) : 0.f;
  }
}

__global__ void k_wedge(float* __restrict__ w, const float* __restrict__ dinv,
                        const int* __restrict__ src, const int* __restrict__ dst, int e) {
  int i = blockIdx.x * 256 + threadIdx.x;
  if (i < e) w[i] = -dinv[src[i]] * dinv[dst[i]];
}

// ---------------- SpMM: out[dst] += w * in[src]  (L2-resident atomics) ----------------
__global__ void k_spmm(float* __restrict__ out, const float* __restrict__ in,
                       const float* __restrict__ wedge, const int* __restrict__ src,
                       const int* __restrict__ dst, int F, int e) {
  int ei = blockIdx.x * 256 + threadIdx.x;
  if (ei >= e) return;
  int c = blockIdx.y * 4;
  float w = wedge[ei];
  int s = src[ei], d = dst[ei];
  const float4 v = *(const float4*)(in + (size_t)s * F + c);
  float* op = out + (size_t)d * F + c;
  atomicAdd(op + 0, w * v.x);
  atomicAdd(op + 1, w * v.y);
  atomicAdd(op + 2, w * v.z);
  atomicAdd(op + 3, w * v.w);
}

// ---------------- weight repack: f32 [K,NOUT] -> bf16 WMMA B-fragment layout ----------
// P[((tn*KS + ks)*32 + lane)*16 + j] = B[ks*32 + (lane>>4)*8 + (j&7) + (j>>3)*16][tn*16 + (lane&15)]
__global__ void k_packB(__bf16* __restrict__ P, const float* __restrict__ B,
                        int K, int NOUT) {
  int i = blockIdx.x * 256 + threadIdx.x;
  int KS = K >> 5;
  int total = (NOUT >> 4) * KS * 512;
  if (i >= total) return;
  int j = i & 15;
  int lane = (i >> 4) & 31;
  int rest = i >> 9;            // tn*KS + ks
  int tn = rest / KS;
  int ks = rest - tn * KS;
  int row = ks * 32 + (lane >> 4) * 8 + (j & 7) + (j >> 3) * 16;
  int col = tn * 16 + (lane & 15);
  P[i] = (__bf16)B[(size_t)row * NOUT + col];
}

// ---------------- WMMA GEMM: C[M,NOUT] += A[M,K] @ B[K,NOUT] ----------------
// One wave = 16 rows x NOUT cols: A fragment loaded/converted once per K-step,
// reused across all NOUT/16 column tiles (packed bf16 B fragments, 32B vector loads).
// A frag (16x32 bf16): lane L holds row L%16; elems 0..7 = K[k0+(L/16)*8 ..],
//                      elems 8..15 = K[k0+16+(L/16)*8 ..]
// C/D (16x16 f32, 8 VGPR): vgpr r = row (L/16)*8 + r, col L%16
template <int K, int NOUT>
__global__ __launch_bounds__(256) void k_gemm_wmma(float* __restrict__ C,
                                                   const float* __restrict__ A,
                                                   const __bf16* __restrict__ Bp, int M) {
  constexpr int TN = NOUT / 16;
  constexpr int KS = K / 32;
  const int gw = (blockIdx.x * 256 + threadIdx.x) >> 5;  // wave id == row tile (uniform)
  const int lane = threadIdx.x & 31;
  if (gw >= (M >> 4)) return;  // wave-uniform: EXEC stays all-ones for WMMA
  const int rc = lane & 15;
  const int hi = lane >> 4;
  const int arow = gw * 16 + rc;
  const size_t c0 = (size_t)(gw * 16 + hi * 8) * NOUT + rc;

  v8f c[TN];
#pragma unroll
  for (int t = 0; t < TN; ++t)
#pragma unroll
    for (int r = 0; r < 8; ++r) c[t][r] = C[c0 + t * 16 + (size_t)r * NOUT];

#pragma unroll
  for (int ks = 0; ks < KS; ++ks) {
    const float4* Ap = (const float4*)(A + (size_t)arow * K + ks * 32 + hi * 8);
    float4 a0 = Ap[0], a1 = Ap[1];  // K = base .. base+7
    float4 a2 = Ap[4], a3 = Ap[5];  // K = base+16 .. base+23
    v16bf a;
    a[0]  = (__bf16)a0.x; a[1]  = (__bf16)a0.y; a[2]  = (__bf16)a0.z; a[3]  = (__bf16)a0.w;
    a[4]  = (__bf16)a1.x; a[5]  = (__bf16)a1.y; a[6]  = (__bf16)a1.z; a[7]  = (__bf16)a1.w;
    a[8]  = (__bf16)a2.x; a[9]  = (__bf16)a2.y; a[10] = (__bf16)a2.z; a[11] = (__bf16)a2.w;
    a[12] = (__bf16)a3.x; a[13] = (__bf16)a3.y; a[14] = (__bf16)a3.z; a[15] = (__bf16)a3.w;

#pragma unroll
    for (int t = 0; t < TN; ++t) {
      v16bf b = *(const v16bf*)(Bp + ((size_t)(t * KS + ks) * 32 + lane) * 16);
      c[t] = __builtin_amdgcn_wmma_f32_16x16x32_bf16(false, a, false, b, (short)0,
                                                     c[t], false, false);
    }
  }

#pragma unroll
  for (int t = 0; t < TN; ++t)
#pragma unroll
    for (int r = 0; r < 8; ++r) C[c0 + t * 16 + (size_t)r * NOUT] = c[t][r];
}

// ---------------- host orchestration ----------------
static inline int cdiv(long long a, long long b) { return (int)((a + b - 1) / b); }

extern "C" void kernel_launch(void* const* d_in, const int* in_sizes, int n_in,
                              void* d_out, int out_size, void* d_ws, size_t ws_size,
                              hipStream_t stream) {
  const int Nn = NNODES, Ee = NEDGES;
  const float* x    = (const float*)d_in[0];
  const int*  srcE  = (const int*)d_in[1];
  const int*  dstE  = (const int*)d_in[2];
  const float* Wb[3]  = {(const float*)d_in[3], (const float*)d_in[7],  (const float*)d_in[11]};
  const float* bb[3]  = {(const float*)d_in[4], (const float*)d_in[8],  (const float*)d_in[12]};
  const float* Wsk[3] = {(const float*)d_in[5], (const float*)d_in[9],  (const float*)d_in[13]};
  const float* bsk[3] = {(const float*)d_in[6], (const float*)d_in[10], (const float*)d_in[14]};
  const float* Wmix = (const float*)d_in[15];
  const float* bmix = (const float*)d_in[16];
  float* out = (float*)d_out;

  // workspace layout (floats, then bf16 pack area; all offsets 32B-aligned)
  float* ws    = (float*)d_ws;
  float* wedge = ws;                          // E
  float* dinv  = wedge + Ee;                  // N
  float* T0    = dinv + Nn;                   // N*160 (also hcat)
  float* T1    = T0 + (size_t)Nn * 160;       // N*160 (also P for final prop)
  float* T2    = T1 + (size_t)Nn * 160;       // N*96  (P buffer for recurrence)
  float* outB  = T2 + (size_t)Nn * 96;        // N*64
  float* outS  = outB + (size_t)Nn * 64;      // N*64
  float* hbuf  = outS + (size_t)Nn * 64;      // N*64
  __bf16* pWb[3], *pWs[3], *pWmix;
  {
    __bf16* pk = (__bf16*)(hbuf + (size_t)Nn * 64);
    pWb[0] = pk;               pk += 6 * 96 * 64;
    pWs[0] = pk;               pk += 96 * 64;
    pWb[1] = pk;               pk += 6 * 64 * 64;
    pWs[1] = pk;               pk += 64 * 64;
    pWb[2] = pk;               pk += 6 * 64 * 64;
    pWs[2] = pk;               pk += 64 * 64;
    pWmix  = pk;
  }

  auto zero = [&](float* p, long long n) {
    k_zero<<<cdiv(n, 256), 256, 0, stream>>>(p, n);
  };
  auto spmm = [&](float* O, const float* I, int F) {
    zero(O, (long long)Nn * F);
    dim3 g(cdiv(Ee, 256), F / 4);
    k_spmm<<<g, 256, 0, stream>>>(O, I, wedge, srcE, dstE, F, Ee);
  };
  auto pack = [&](__bf16* P, const float* B, int K, int NOUT) {
    k_packB<<<cdiv((long long)K * NOUT, 256), 256, 0, stream>>>(P, B, K, NOUT);
  };

  // ---- pack all weights to bf16 WMMA fragment layout (tiny, once per launch) ----
  for (int l = 0; l < 3; ++l) {
    const int Fi = (l == 0) ? 96 : 64;
    for (int k = 0; k < 6; ++k)
      pack(pWb[l] + (size_t)k * Fi * 64, Wb[l] + (size_t)k * Fi * 64, Fi, 64);
    pack(pWs[l], Wsk[l], Fi, 64);
  }
  pack(pWmix, Wmix, 160, 48);
  pack(pWmix + 160 * 48, Wmix + 160 * 48, 160, 48);

  // ---- edge weights: w = -dinv[src]*dinv[dst], deg counted over src ----
  zero(dinv, Nn);
  k_deg<<<cdiv(Ee, 256), 256, 0, stream>>>(dinv, srcE, Ee);
  k_dinv<<<cdiv(Nn, 256), 256, 0, stream>>>(dinv, Nn);
  k_wedge<<<cdiv(Ee, 256), 256, 0, stream>>>(wedge, dinv, srcE, dstE, Ee);

  const long long waveThreads = (long long)(Nn / 16) * 32;  // one wave per 16-row tile
  const int gemmBlocks = cdiv(waveThreads, 256);

  const float* h = x;
  for (int l = 0; l < 3; ++l) {
    const int Fi = (l == 0) ? 96 : 64;
    auto gemm64 = [&](float* C, const float* A, const __bf16* B) {
      if (Fi == 96)
        k_gemm_wmma<96, 64><<<gemmBlocks, 256, 0, stream>>>(C, A, B, Nn);
      else
        k_gemm_wmma<64, 64><<<gemmBlocks, 256, 0, stream>>>(C, A, B, Nn);
    };

    // outB = bb + T0@Wb[0];  outS = bs + h@Ws  (h still pristine here)
    k_bias_init<<<cdiv((long long)Nn * 64, 256), 256, 0, stream>>>(outB, bb[l], 64, (long long)Nn * 64);
    gemm64(outB, h, pWb[l]);
    k_bias_init<<<cdiv((long long)Nn * 64, 256), 256, 0, stream>>>(outS, bsk[l], 64, (long long)Nn * 64);
    gemm64(outS, h, pWs[l]);

    // T0 = h (mutable copy for recurrence); T1 = prop(T0)
    k_copy<<<cdiv((long long)Nn * Fi, 256), 256, 0, stream>>>(T0, h, (long long)Nn * Fi);
    spmm(T1, h, Fi);
    gemm64(outB, T1, pWb[l] + (size_t)1 * Fi * 64);

    float* tp = T0;  // T_{k-2}
    float* tc = T1;  // T_{k-1}
    for (int k = 2; k < 6; ++k) {
      spmm(T2, tc, Fi);                                   // P = prop(T_{k-1})
      k_combine<<<cdiv((long long)Nn * Fi, 256), 256, 0, stream>>>(tp, T2, (long long)Nn * Fi);
      float* t = tp; tp = tc; tc = t;                     // tc now holds T_k
      gemm64(outB, tc, pWb[l] + (size_t)k * Fi * 64);
    }

    // h = relu(outB) + outS
    k_relu_skip<<<cdiv((long long)Nn * 64, 256), 256, 0, stream>>>(hbuf, outB, outS, (long long)Nn * 64);
    h = hbuf;
  }

  // ---- final mix: hcat=[h|x] (160), out = bmix + hcat@Wmix0 + prop(hcat)@Wmix1 ----
  k_concat<<<cdiv((long long)Nn * 160, 256), 256, 0, stream>>>(T0, hbuf, x, Nn);
  k_bias_init<<<cdiv((long long)Nn * 48, 256), 256, 0, stream>>>(out, bmix, 48, (long long)Nn * 48);
  k_gemm_wmma<160, 48><<<gemmBlocks, 256, 0, stream>>>(out, T0, pWmix, Nn);
  spmm(T1, T0, 160);
  k_gemm_wmma<160, 48><<<gemmBlocks, 256, 0, stream>>>(out, T1, pWmix + 160 * 48, Nn);
}
